// PosNetEncoder_3384434230183
// MI455X (gfx1250) — compile-verified, matrix operand order checked by reading, not run on hardware
//
#include <hip/hip_runtime.h>
#include <math.h>
#include <stdint.h>

// ---------------- problem constants (fixed by the reference) ----------------
static constexpr int Bb  = 2;
static constexpr int Ii  = 1024;
static constexpr int Dd  = 1024;
static constexpr int Hh  = 16;
static constexpr int DH  = 64;
static constexpr int Jj  = 1024;
static constexpr int FF  = 4096;
static constexpr int KCL = 64;          // clamp radius
static constexpr int Rr  = 2 * KCL + 1; // 129
static constexpr int RP  = 132;         // padded arf row stride
static constexpr long long NROW = (long long)Bb * Ii;  // 2048

// ---------------- bf16 helpers ----------------
typedef __bf16 bf16;
typedef bf16  bf16x16 __attribute__((ext_vector_type(16)));
typedef float floatx8 __attribute__((ext_vector_type(8)));

__device__ inline bf16 f2bf(float f) {
  unsigned int u = __builtin_bit_cast(unsigned int, f);
  unsigned int r = (u + 0x7FFFu + ((u >> 16) & 1u)) >> 16;  // RNE
  return __builtin_bit_cast(bf16, (unsigned short)r);
}
__device__ inline unsigned short bf_bits(float f) {
  unsigned int u = __builtin_bit_cast(unsigned int, f);
  return (unsigned short)((u + 0x7FFFu + ((u >> 16) & 1u)) >> 16);
}

__device__ inline float gelu_exact(float x) {
  return 0.5f * x * (1.0f + erff(x * 0.70710678118654752440f));
}

// ---------------- CDNA5 async global->LDS staging ----------------
// Each lane copies 16B; 128 threads stage a 64x64 bf16 tile in 4 issues/lane.
// Tracked on ASYNCcnt; fenced with s_wait_asynccnt + workgroup barrier.
__device__ inline void async_tile64(bf16* dstLds, const bf16* src, long long ldSrc) {
  int t   = threadIdx.x;          // 128 threads
  int col = (t & 7) * 8;          // 8 bf16 = 16B per lane
  int row = t >> 3;               // 16 rows per round
  uint32_t base = (uint32_t)(size_t)dstLds;
#pragma unroll
  for (int rr = 0; rr < 4; ++rr) {
    int m = row + rr * 16;
    uint32_t lds = base + (uint32_t)((m * 64 + col) * 2);
    const bf16* g = src + (long long)m * ldSrc + col;
    asm volatile("global_load_async_to_lds_b128 %0, %1, off"
                 :: "v"(lds), "v"(g) : "memory");
  }
}
__device__ inline void async_wait0() {
  asm volatile("s_wait_asynccnt 0" ::: "memory");
}

// ---------------- WMMA fragment helpers (wave32 layouts, ISA 7.12.2) -------
__device__ inline bf16x16 load_frag16(const bf16* base, int ld) {
  int lane = threadIdx.x & 31;
  const bf16* row = base + (lane & 15) * ld;
  int hi = (lane >> 4) << 3;  // 0 or 8
  bf16x16 f;
#pragma unroll
  for (int p = 0; p < 8; ++p) {
    int k = ((p & 4) << 2) + hi + ((p & 3) << 1);
    f[2 * p]     = row[k];
    f[2 * p + 1] = row[k + 1];
  }
  return f;
}
__device__ inline floatx8 wmma_bf16(bf16x16 a, bf16x16 b, floatx8 c) {
  return __builtin_amdgcn_wmma_f32_16x16x32_bf16(false, a, false, b,
                                                 (short)0, c, false, false);
}

// ---------------- one-time converts ----------------
__global__ __launch_bounds__(256) void conv_kernel(const float* __restrict__ src,
                                                   bf16* __restrict__ dst,
                                                   long long n4) {
  long long i = (long long)blockIdx.x * 256 + threadIdx.x;
  if (i >= n4) return;
  float4 v = ((const float4*)src)[i];
  uint64_t r = (uint64_t)bf_bits(v.x)
             | ((uint64_t)bf_bits(v.y) << 16)
             | ((uint64_t)bf_bits(v.z) << 32)
             | ((uint64_t)bf_bits(v.w) << 48);
  ((uint64_t*)dst)[i] = r;
}

// dst[n*K + k] = src[k*N + n]  (weight transpose to [N][K] bf16)
__global__ __launch_bounds__(256) void convT_kernel(const float* __restrict__ src,
                                                    bf16* __restrict__ dst,
                                                    int K, int N) {
  long long i = (long long)blockIdx.x * 256 + threadIdx.x;
  if (i >= (long long)K * N) return;
  int k = (int)(i / N), n = (int)(i % N);
  dst[(long long)n * K + k] = f2bf(src[i]);
}

// ---------------- LayerNorm -> bf16 (optionally transposed) ----------------
__global__ __launch_bounds__(256) void ln_kernel(const float* __restrict__ x,
                                                 const float* __restrict__ gamma,
                                                 const float* __restrict__ beta,
                                                 bf16* __restrict__ outB,
                                                 long long ldb, int trans, int n) {
  __shared__ float s1[256], s2[256];
  long long row = blockIdx.x;
  const float* xr = x + row * n;
  float sum = 0.f, sq = 0.f;
  for (int c = threadIdx.x; c < n; c += 256) {
    float v = xr[c];
    sum += v; sq += v * v;
  }
  s1[threadIdx.x] = sum; s2[threadIdx.x] = sq;
  __syncthreads();
  for (int off = 128; off > 0; off >>= 1) {
    if ((int)threadIdx.x < off) {
      s1[threadIdx.x] += s1[threadIdx.x + off];
      s2[threadIdx.x] += s2[threadIdx.x + off];
    }
    __syncthreads();
  }
  float mean = s1[0] / n;
  float var  = s2[0] / n - mean * mean;
  float rstd = rsqrtf(var + 1e-5f);
  for (int c = threadIdx.x; c < n; c += 256) {
    float y = (xr[c] - mean) * rstd * gamma[c] + beta[c];
    if (trans) outB[(long long)c * ldb + row] = f2bf(y);
    else       outB[row * ldb + c]            = f2bf(y);
  }
}

// ---------------- generic bf16 WMMA GEMM ----------------
// C[M,N] = epi( A[M,K] @ BT[N,K]^T ), A/BT bf16, async-staged through LDS.
// epi: +bias[n] -> act(0 none,1 gelu,2 relu) -> *gate[m,n] -> +resid[m,n]
__global__ __launch_bounds__(128) void gemm_bf16_kernel(
    const bf16* __restrict__ A, long long lda, long long aZ,
    const bf16* __restrict__ BT, long long ldbt, long long bZ,
    const float* __restrict__ bias,
    const float* __restrict__ gate, long long ldg, long long gZ,
    const float* __restrict__ resid, long long ldr,
    float* __restrict__ outF, bf16* __restrict__ outB,
    long long ldo, long long oZ,
    int M, int N, int K, int act) {
  __shared__ bf16 As[64][64];
  __shared__ bf16 Bs[64][64];
  long long z = blockIdx.z;
  A += z * aZ; BT += z * bZ;
  if (outF) outF += z * oZ;
  if (outB) outB += z * oZ;
  if (gate) gate += z * gZ;
  int m0 = blockIdx.y * 64, n0 = blockIdx.x * 64;
  int tid = threadIdx.x, wave = tid >> 5, lane = tid & 31;
  int wm = (wave >> 1) * 32, wn = (wave & 1) * 32;
  floatx8 acc[2][2] = {};
  for (int kk0 = 0; kk0 < K; kk0 += 64) {
    async_tile64(&As[0][0], A  + (long long)m0 * lda  + kk0, lda);
    async_tile64(&Bs[0][0], BT + (long long)n0 * ldbt + kk0, ldbt);
    async_wait0();
    __syncthreads();
#pragma unroll
    for (int kk = 0; kk < 64; kk += 32) {
      bf16x16 af[2], bfr[2];
      af[0]  = load_frag16(&As[wm][kk], 64);
      af[1]  = load_frag16(&As[wm + 16][kk], 64);
      bfr[0] = load_frag16(&Bs[wn][kk], 64);
      bfr[1] = load_frag16(&Bs[wn + 16][kk], 64);
#pragma unroll
      for (int tm = 0; tm < 2; ++tm)
#pragma unroll
        for (int tn = 0; tn < 2; ++tn)
          acc[tm][tn] = wmma_bf16(af[tm], bfr[tn], acc[tm][tn]);
    }
    __syncthreads();
  }
  int hi = (lane >> 4) << 3, nlo = lane & 15;
#pragma unroll
  for (int tm = 0; tm < 2; ++tm)
#pragma unroll
    for (int tn = 0; tn < 2; ++tn)
#pragma unroll
      for (int r = 0; r < 8; ++r) {
        long long m = m0 + wm + tm * 16 + hi + r;
        long long n = n0 + wn + tn * 16 + nlo;
        float v = acc[tm][tn][r];
        if (bias)  v += bias[n];
        if (act == 1) v = gelu_exact(v);
        else if (act == 2) v = fmaxf(v, 0.f);
        if (gate)  v *= gate[m * ldg + n];
        if (resid) v += resid[m * ldr + n];
        if (outF)  outF[m * ldo + n] = v;
        if (outB)  outB[m * ldo + n] = f2bf(v);
      }
}

// ---------------- arf[b,h,i,r] = q2_h[i,:] . rel_h[r,:] ----------------
__global__ __launch_bounds__(128) void arf_kernel(const bf16* __restrict__ q2,
                                                  const bf16* __restrict__ relb,
                                                  float* __restrict__ arf) {
  __shared__ bf16 As[64][64];
  __shared__ bf16 Bs[144][64];
  int zz = blockIdx.y;  // b*H + h
  int b = zz >> 4, h = zz & 15;
  int i0 = blockIdx.x * 64;
  const bf16* Ap = q2 + (long long)b * Ii * Dd + h * DH;
  const bf16* Rp = relb + h * DH;
  float* outp = arf + (long long)zz * Ii * RP;
  int tid = threadIdx.x;
  async_tile64(&As[0][0], Ap + (long long)i0 * Dd, Dd);
  {
    int col = (tid & 7) * 8, row = tid >> 3;
    uint32_t base = (uint32_t)(size_t)&Bs[0][0];
#pragma unroll
    for (int rr = 0; rr < 9; ++rr) {
      int m = row + rr * 16;  // 0..143
      if (m < Rr) {
        uint32_t lds = base + (uint32_t)((m * 64 + col) * 2);
        const bf16* g = Rp + (long long)m * Dd + col;
        asm volatile("global_load_async_to_lds_b128 %0, %1, off"
                     :: "v"(lds), "v"(g) : "memory");
      } else {
        bf16* zp = &Bs[0][0] + m * 64 + col;
#pragma unroll
        for (int q = 0; q < 8; ++q) zp[q] = f2bf(0.f);
      }
    }
  }
  async_wait0();
  __syncthreads();
  int wave = tid >> 5, lane = tid & 31;
  int wrow = wave * 16;
  floatx8 acc[9] = {};
#pragma unroll
  for (int kk = 0; kk < 64; kk += 32) {
    bf16x16 af = load_frag16(&As[wrow][kk], 64);
#pragma unroll
    for (int t = 0; t < 9; ++t) {
      bf16x16 bfr = load_frag16(&Bs[t * 16][kk], 64);
      acc[t] = wmma_bf16(af, bfr, acc[t]);
    }
  }
  int hi = (lane >> 4) << 3, nlo = lane & 15;
#pragma unroll
  for (int t = 0; t < 9; ++t)
#pragma unroll
    for (int r = 0; r < 8; ++r) {
      int i = i0 + wrow + hi + r;
      int n = t * 16 + nlo;
      if (n < RP) outp[(long long)i * RP + n] = acc[t][r];
    }
}

// ------------- scores: (q1.k^T + gather(arf)) / 8 -> a (pre-softmax) -------
__global__ __launch_bounds__(128) void scores_kernel(const bf16* __restrict__ q1,
                                                     const bf16* __restrict__ kb,
                                                     const float* __restrict__ arf,
                                                     float* __restrict__ a_out) {
  __shared__ bf16 As[64][64];
  __shared__ bf16 Bs[64][64];
  int zz = blockIdx.z, b = zz >> 4, h = zz & 15;
  const bf16* Aq = q1 + (long long)b * Ii * Dd + h * DH;
  const bf16* Kp = kb + (long long)b * Ii * Dd + h * DH;
  const float* arfp = arf + (long long)zz * Ii * RP;
  float* outp = a_out + (long long)zz * Ii * Jj;
  int i0 = blockIdx.y * 64, j0 = blockIdx.x * 64;
  async_tile64(&As[0][0], Aq + (long long)i0 * Dd, Dd);
  async_tile64(&Bs[0][0], Kp + (long long)j0 * Dd, Dd);
  async_wait0();
  __syncthreads();
  int tid = threadIdx.x, wave = tid >> 5, lane = tid & 31;
  int wm = (wave >> 1) * 32, wn = (wave & 1) * 32;
  floatx8 acc[2][2] = {};
#pragma unroll
  for (int kk = 0; kk < 64; kk += 32) {
    bf16x16 af[2], bfr[2];
    af[0]  = load_frag16(&As[wm][kk], 64);
    af[1]  = load_frag16(&As[wm + 16][kk], 64);
    bfr[0] = load_frag16(&Bs[wn][kk], 64);
    bfr[1] = load_frag16(&Bs[wn + 16][kk], 64);
#pragma unroll
    for (int tm = 0; tm < 2; ++tm)
#pragma unroll
      for (int tn = 0; tn < 2; ++tn)
        acc[tm][tn] = wmma_bf16(af[tm], bfr[tn], acc[tm][tn]);
  }
  int hi = (lane >> 4) << 3, nlo = lane & 15;
#pragma unroll
  for (int tm = 0; tm < 2; ++tm)
#pragma unroll
    for (int tn = 0; tn < 2; ++tn)
#pragma unroll
      for (int r = 0; r < 8; ++r) {
        int i  = i0 + wm + tm * 16 + hi + r;
        int jj = j0 + wn + tn * 16 + nlo;
        int rel = jj - i;
        rel = rel < -KCL ? -KCL : (rel > KCL ? KCL : rel);
        rel += KCL;
        float s = (acc[tm][tn][r] + arfp[(long long)i * RP + rel]) * 0.125f;
        outp[(long long)i * Jj + jj] = s;
      }
}

// ---------------- softmax over J: f32 in place + bf16 copy ----------------
__global__ __launch_bounds__(256) void softmax_kernel(float* __restrict__ a,
                                                      bf16* __restrict__ ab) {
  __shared__ float red[256];
  long long row = blockIdx.x;
  float* ar = a + row * (long long)Jj;
  bf16* abr = ab + row * (long long)Jj;
  float vals[4];
  float mx = -INFINITY;
#pragma unroll
  for (int t = 0; t < 4; ++t) {
    vals[t] = ar[threadIdx.x + t * 256];
    mx = fmaxf(mx, vals[t]);
  }
  red[threadIdx.x] = mx;
  __syncthreads();
  for (int off = 128; off > 0; off >>= 1) {
    if ((int)threadIdx.x < off)
      red[threadIdx.x] = fmaxf(red[threadIdx.x], red[threadIdx.x + off]);
    __syncthreads();
  }
  mx = red[0];
  __syncthreads();
  float sum = 0.f;
#pragma unroll
  for (int t = 0; t < 4; ++t) {
    vals[t] = expf(vals[t] - mx);
    sum += vals[t];
  }
  red[threadIdx.x] = sum;
  __syncthreads();
  for (int off = 128; off > 0; off >>= 1) {
    if ((int)threadIdx.x < off)
      red[threadIdx.x] += red[threadIdx.x + off];
    __syncthreads();
  }
  float inv = 1.f / red[0];
#pragma unroll
  for (int t = 0; t < 4; ++t) {
    float p = vals[t] * inv;
    ar[threadIdx.x + t * 256]  = p;
    abr[threadIdx.x + t * 256] = f2bf(p);
  }
}

// ---------------- launch ----------------
extern "C" void kernel_launch(void* const* d_in, const int* in_sizes, int n_in,
                              void* d_out, int out_size, void* d_ws, size_t ws_size,
                              hipStream_t stream) {
  const float* x     = (const float*)d_in[0];
  const float* j     = (const float*)d_in[1];
  const float* ln1_g = (const float*)d_in[2];
  const float* ln1_b = (const float*)d_in[3];
  const float* Wv_w  = (const float*)d_in[4];
  const float* Wv_b  = (const float*)d_in[5];
  const float* Wg_w  = (const float*)d_in[6];
  const float* Wg_b  = (const float*)d_in[7];
  const float* lnv_g = (const float*)d_in[8];
  const float* lnv_b = (const float*)d_in[9];
  const float* Wq1_w = (const float*)d_in[10];
  const float* Wq1_b = (const float*)d_in[11];
  const float* Wk_w  = (const float*)d_in[12];
  const float* Wk_b  = (const float*)d_in[13];
  const float* Wq2_w = (const float*)d_in[14];
  const float* Wq2_b = (const float*)d_in[15];
  const float* rel   = (const float*)d_in[16];
  const float* Wo_w  = (const float*)d_in[17];
  const float* Wo_b  = (const float*)d_in[18];
  const float* ln2_g = (const float*)d_in[19];
  const float* ln2_b = (const float*)d_in[20];
  const float* ff1_w = (const float*)d_in[21];
  const float* ff1_b = (const float*)d_in[22];
  const float* ff2_w = (const float*)d_in[23];
  const float* ff2_b = (const float*)d_in[24];
  (void)n_in; (void)in_sizes; (void)out_size; (void)ws_size;

  float* outp  = (float*)d_out;                 // [B,I,D]
  float* a_out = (float*)d_out + NROW * Dd;     // [B,H,I,J]

  // ---- workspace carve-up (all chunks 16B-aligned) ----
  char* p = (char*)d_ws;
  auto allocF = [&](long long n) { float* r = (float*)p; p += n * 4; return r; };
  auto allocB = [&](long long n) { bf16* r = (bf16*)p; p += n * 2; return r; };
  float* vpre = allocF(NROW * Dd);
  float* gbuf = allocF(NROW * Dd);
  float* x1   = allocF(NROW * Dd);
  float* arfb = allocF((long long)Bb * Hh * Ii * RP);
  bf16* xn_bf  = allocB(NROW * Dd);
  bf16* j_bf   = allocB(NROW * Dd);
  bf16* q1_bf  = allocB(NROW * Dd);
  bf16* k_bf   = allocB(NROW * Dd);
  bf16* q2_bf  = allocB(NROW * Dd);
  bf16* vT_bf  = allocB((long long)Dd * NROW);         // [D][B*I]
  bf16* o_bf   = allocB(NROW * Dd);
  bf16* x1n_bf = allocB(NROW * Dd);
  bf16* h_bf   = allocB(NROW * FF);
  bf16* a_bf   = allocB((long long)Bb * Hh * Ii * Jj); // [B,H,I,J]
  bf16* WvT  = allocB((long long)Dd * Dd);
  bf16* WgT  = allocB((long long)Dd * Dd);
  bf16* Wq1T = allocB((long long)Dd * Dd);
  bf16* WkT  = allocB((long long)Dd * Dd);
  bf16* Wq2T = allocB((long long)Dd * Dd);
  bf16* WoT  = allocB((long long)Dd * Dd);
  bf16* ff1T = allocB((long long)Dd * FF);             // [FF][D]
  bf16* ff2T = allocB((long long)FF * Dd);             // [D][FF]
  bf16* rel_bf = allocB((long long)Rr * Dd);

  auto gemm = [&](const bf16* A, long long lda, long long aZ,
                  const bf16* BT, long long ldbt, long long bZ,
                  const float* bias,
                  const float* gate, long long ldg, long long gZ,
                  const float* resid, long long ldr,
                  float* oF, bf16* oB, long long ldo, long long oZ,
                  int M, int N, int K, int act, int Z) {
    dim3 grid(N / 64, M / 64, Z);
    gemm_bf16_kernel<<<grid, dim3(128), 0, stream>>>(
        A, lda, aZ, BT, ldbt, bZ, bias, gate, ldg, gZ, resid, ldr,
        oF, oB, ldo, oZ, M, N, K, act);
  };
  auto convT = [&](const float* src, bf16* dst, int K, int N) {
    long long tot = (long long)K * N;
    convT_kernel<<<dim3((unsigned)((tot + 255) / 256)), dim3(256), 0, stream>>>(src, dst, K, N);
  };
  auto conv = [&](const float* src, bf16* dst, long long cnt) {
    long long n4 = cnt / 4;
    conv_kernel<<<dim3((unsigned)((n4 + 255) / 256)), dim3(256), 0, stream>>>(src, dst, n4);
  };

  // ---- 0) one-time bf16 conversions ----
  convT(Wv_w, WvT, Dd, Dd);   convT(Wg_w, WgT, Dd, Dd);
  convT(Wq1_w, Wq1T, Dd, Dd); convT(Wk_w, WkT, Dd, Dd);
  convT(Wq2_w, Wq2T, Dd, Dd); convT(Wo_w, WoT, Dd, Dd);
  convT(ff1_w, ff1T, Dd, FF); convT(ff2_w, ff2T, FF, Dd);
  conv(j, j_bf, NROW * Dd);
  conv(rel, rel_bf, (long long)Rr * Dd);

  // ---- 1) xn = LN1(x) -> bf16 ----
  ln_kernel<<<dim3((unsigned)NROW), dim3(256), 0, stream>>>(x, ln1_g, ln1_b, xn_bf, Dd, 0, Dd);
  // ---- 2) v-pre = gelu(xn@Wv+b) -> f32 ----
  gemm(xn_bf, Dd, 0, WvT, Dd, 0, Wv_b, nullptr, 0, 0, nullptr, 0,
       vpre, nullptr, Dd, 0, (int)NROW, Dd, Dd, 1, 1);
  // ---- 3) v = LNv(v-pre) -> vT bf16 [D][B*I] ----
  ln_kernel<<<dim3((unsigned)NROW), dim3(256), 0, stream>>>(vpre, lnv_g, lnv_b, vT_bf, NROW, 1, Dd);
  // ---- 4) g = gelu(xn@Wg+b) -> f32 ----
  gemm(xn_bf, Dd, 0, WgT, Dd, 0, Wg_b, nullptr, 0, 0, nullptr, 0,
       gbuf, nullptr, Dd, 0, (int)NROW, Dd, Dd, 1, 1);
  // ---- 5-7) q1, k, q2 -> bf16 ----
  gemm(j_bf, Dd, 0, Wq1T, Dd, 0, Wq1_b, nullptr, 0, 0, nullptr, 0,
       nullptr, q1_bf, Dd, 0, (int)NROW, Dd, Dd, 0, 1);
  gemm(j_bf, Dd, 0, WkT, Dd, 0, Wk_b, nullptr, 0, 0, nullptr, 0,
       nullptr, k_bf, Dd, 0, (int)NROW, Dd, Dd, 0, 1);
  gemm(j_bf, Dd, 0, Wq2T, Dd, 0, Wq2_b, nullptr, 0, 0, nullptr, 0,
       nullptr, q2_bf, Dd, 0, (int)NROW, Dd, Dd, 0, 1);
  // ---- 8) arf ----
  arf_kernel<<<dim3(Ii / 64, Bb * Hh), dim3(128), 0, stream>>>(q2_bf, rel_bf, arfb);
  // ---- 9) scores -> a_out (pre-softmax) ----
  scores_kernel<<<dim3(Jj / 64, Ii / 64, Bb * Hh), dim3(128), 0, stream>>>(
      q1_bf, k_bf, arfb, a_out);
  // ---- 10) softmax (f32 in d_out, bf16 copy for the a@v GEMM) ----
  softmax_kernel<<<dim3((unsigned)((long long)Bb * Hh * Ii)), dim3(256), 0, stream>>>(a_out, a_bf);
  // ---- 11) o_h = (a @ v_h) * g_h -> o_bf ; z iterates heads ----
  for (int b = 0; b < Bb; ++b) {
    gemm(a_bf + (long long)b * Hh * Ii * Jj, Jj, (long long)Ii * Jj,
         vT_bf + (long long)b * Ii, NROW, (long long)DH * NROW,
         nullptr,
         gbuf + (long long)b * Ii * Dd, Dd, DH,
         nullptr, 0,
         nullptr, o_bf + (long long)b * Ii * Dd, Dd, DH,
         Ii, DH, Jj, 0, Hh);
  }
  // ---- 12) x1 = o@Wo + b + x -> f32 ----
  gemm(o_bf, Dd, 0, WoT, Dd, 0, Wo_b, nullptr, 0, 0, x, Dd,
       x1, nullptr, Dd, 0, (int)NROW, Dd, Dd, 0, 1);
  // ---- 13) x1n = LN2(x1) -> bf16 ----
  ln_kernel<<<dim3((unsigned)NROW), dim3(256), 0, stream>>>(x1, ln2_g, ln2_b, x1n_bf, Dd, 0, Dd);
  // ---- 14) h = relu(x1n@ff1+b) -> bf16 ----
  gemm(x1n_bf, Dd, 0, ff1T, Dd, 0, ff1_b, nullptr, 0, 0, nullptr, 0,
       nullptr, h_bf, FF, 0, (int)NROW, FF, Dd, 2, 1);
  // ---- 15) out = h@ff2 + b + x1 ----
  gemm(h_bf, FF, 0, ff2T, FF, 0, ff2_b, nullptr, 0, 0, x1, Dd,
       outp, nullptr, Dd, 0, (int)NROW, Dd, FF, 0, 1);
}
